// MultiHeadAttention_60155311948376
// MI455X (gfx1250) — matrix operand order repaired and checked by reading, not compiled, not run on hardware
//
#include <hip/hip_runtime.h>
#include <hip/hip_bf16.h>

// ---------------------------------------------------------------------------
// MHA for gfx1250 (MI455X): B=2, S=2048, D=1024, H=16, DH=64, causal.
// Compute-bound (~70 GFLOP vs ~90MB traffic) -> everything goes through
// v_wmma_f32_16x16x32_f16 with fp32 accumulation.
// ---------------------------------------------------------------------------

typedef __attribute__((ext_vector_type(16))) _Float16 v16h;
typedef __attribute__((ext_vector_type(8)))  _Float16 v8h;
typedef __attribute__((ext_vector_type(8)))  float    v8f;

#define S_LEN  2048
#define DMODEL 1024
#define NHEAD  16
#define DHEAD  64

static __device__ __forceinline__ v16h cat8(v8h lo, v8h hi) {
  return __builtin_shufflevector(lo, hi, 0,1,2,3,4,5,6,7,8,9,10,11,12,13,14,15);
}

// 8 contiguous fp32 -> 8 f16 (two b128 loads + v_cvt_pk)
static __device__ __forceinline__ v8h load8f_cvt(const float* p) {
  const float4* q = (const float4*)p;
  float4 a = q[0], b = q[1];
  v8h r;
  r[0]=(_Float16)a.x; r[1]=(_Float16)a.y; r[2]=(_Float16)a.z; r[3]=(_Float16)a.w;
  r[4]=(_Float16)b.x; r[5]=(_Float16)b.y; r[6]=(_Float16)b.z; r[7]=(_Float16)b.w;
  return r;
}
static __device__ __forceinline__ v16h load16f_cvt(const float* p) {
  return cat8(load8f_cvt(p), load8f_cvt(p + 8));
}

static __device__ __forceinline__ v8f wmma16(v16h a, v16h b, v8f c) {
  // (neg_a, A, neg_b, B, c_mod, C, reuse_a, reuse_b)
  return __builtin_amdgcn_wmma_f32_16x16x32_f16(false, a, false, b, (short)0, c,
                                                false, false);
}

// ---------------------------------------------------------------------------
// Projection: out[m,n] = sum_k X[m,k]*W[n,k] + bias[n]   (torch Linear)
// X fp32 [4096,1024], W fp32 [1024,1024], out f16.
// vLayout==0: out[b,h,s,dh] ; vLayout==1: out[b,h,dh,s] (transposed, for V)
// Block = 256 threads (8 waves), block tile 128x128, wave tile 64x32.
// ---------------------------------------------------------------------------
__global__ __launch_bounds__(256) void proj_f16_kernel(
    const float* __restrict__ X, const float* __restrict__ W,
    const float* __restrict__ bias, _Float16* __restrict__ out, int vLayout)
{
  const int lane = threadIdx.x & 31;
  const int w    = threadIdx.x >> 5;
  const int wm   = w & 1, wn = w >> 1;
  const int half = lane >> 4;          // which 16-lane half
  const int l16  = lane & 15;
  const int mbase = blockIdx.x * 128 + wm * 64;
  const int nbase = blockIdx.y * 128 + wn * 32;

  v8f acc[4][2];
  #pragma unroll
  for (int nt = 0; nt < 2; ++nt) {
    float bv = bias[nbase + nt * 16 + l16];   // C layout: lane%16 == column
    #pragma unroll
    for (int mt = 0; mt < 4; ++mt)
      #pragma unroll
      for (int r = 0; r < 8; ++r)
        acc[mt][nt][r] = bv;
  }

  for (int kb = 0; kb < DMODEL; kb += 32) {
    // A fragments: contiguous order: a[i] = X[m][kb + 16h + i]
    // (h = lane/16, row m = lane%16)
    v16h a[4];
    #pragma unroll
    for (int mt = 0; mt < 4; ++mt) {
      const float* xp = X + (size_t)(mbase + mt * 16 + l16) * DMODEL + kb;
      a[mt] = load16f_cvt(xp + 16 * half);
    }
    // B fragments: B[k][n] = W[n][k]; b[i] = W[n][kb + 16h + i] (contiguous)
    v16h b[2];
    #pragma unroll
    for (int nt = 0; nt < 2; ++nt) {
      const float* wp = W + (size_t)(nbase + nt * 16 + l16) * DMODEL + kb + 16 * half;
      b[nt] = load16f_cvt(wp);
    }
    #pragma unroll
    for (int mt = 0; mt < 4; ++mt)
      #pragma unroll
      for (int nt = 0; nt < 2; ++nt)
        acc[mt][nt] = wmma16(a[mt], b[nt], acc[mt][nt]);
  }

  // Store: C layout lane holds col n = l16, rows r + 8*half
  #pragma unroll
  for (int mt = 0; mt < 4; ++mt)
    #pragma unroll
    for (int nt = 0; nt < 2; ++nt) {
      const int n = nbase + nt * 16 + l16;
      const int head = n >> 6, dh = n & 63;
      #pragma unroll
      for (int r = 0; r < 8; ++r) {
        const int m    = mbase + mt * 16 + r + 8 * half;
        const int bidx = m >> 11;                 // token / S
        const int s    = m & (S_LEN - 1);
        size_t idx = (vLayout == 0)
          ? ((size_t)(bidx * NHEAD + head) * S_LEN + s) * DHEAD + dh
          : ((size_t)(bidx * NHEAD + head) * DHEAD + dh) * S_LEN + s;
        out[idx] = (_Float16)acc[mt][nt][r];
      }
    }
}

// ---------------------------------------------------------------------------
// Causal flash attention. One wave per (b,h, 32-query tile) -> K/V fragments
// are register-reused across the two 16-row query tiles (2x arithmetic
// intensity vs a 16-query wave). Q,K in [B,H,S,DH] f16; V transposed
// [B,H,DH,S] f16 so PV B-frags are contiguous 32B loads. P goes through LDS
// to convert C-fragment -> A-fragment layout (same-wave, s_wait_dscnt only).
// Next key block is prefetched with global_prefetch_b8.
// ---------------------------------------------------------------------------
__global__ __launch_bounds__(128) void attn_flash_kernel(
    const _Float16* __restrict__ Q, const _Float16* __restrict__ K,
    const _Float16* __restrict__ Vt, _Float16* __restrict__ ctxOut)
{
  __shared__ __attribute__((aligned(32))) _Float16 Plds[4][2][16][32];

  const int lane = threadIdx.x & 31;
  const int w    = threadIdx.x >> 5;
  const int half = lane >> 4, l16 = lane & 15;
  const int unit  = blockIdx.x * 4 + w;   // 2048 units = B*H*(S/32)
  const int qt    = unit & 63;
  const int bh    = unit >> 6;
  const int qbase = qt * 32;

  const _Float16* Qb = Q  + (size_t)bh * S_LEN * DHEAD;
  const _Float16* Kb = K  + (size_t)bh * S_LEN * DHEAD;
  const _Float16* Vb = Vt + (size_t)bh * DHEAD * S_LEN;

  // Q A-fragments: 2 query tiles x (K=DH=64 -> two 16x32 fragments)
  v16h aq[2][2];
  #pragma unroll
  for (int qq = 0; qq < 2; ++qq) {
    const _Float16* qrow = Qb + (size_t)(qbase + 16 * qq + l16) * DHEAD;
    #pragma unroll
    for (int c = 0; c < 2; ++c) {
      aq[qq][c] = *(const v16h*)(qrow + 32 * c + 16 * half);
    }
  }

  float rmax[2][8], rsum[2][8];
  v8f ctx[2][4];
  #pragma unroll
  for (int qq = 0; qq < 2; ++qq) {
    #pragma unroll
    for (int r = 0; r < 8; ++r) { rmax[qq][r] = -3.0e38f; rsum[qq][r] = 0.0f; }
    #pragma unroll
    for (int t = 0; t < 4; ++t)
      #pragma unroll
      for (int r = 0; r < 8; ++r) ctx[qq][t][r] = 0.0f;
  }

  const float scale = 0.125f;           // DH^-0.5
  const int   nblk  = (qbase >> 5) + 1; // causal: keys <= qbase+31

  for (int jb = 0; jb < nblk; ++jb) {
    const int kb = jb * 32;

    // ---- prefetch next key block (global_prefetch_b8) ----
    if (jb + 1 < nblk) {
      __builtin_prefetch(Kb + (size_t)(kb + 32 + lane) * DHEAD, 0, 3);
      __builtin_prefetch(Vb + (size_t)lane * S_LEN + kb + 32, 0, 3);
      __builtin_prefetch(Vb + (size_t)(lane + 32) * S_LEN + kb + 32, 0, 3);
    }

    // ---- K B-fragments (shared by both query tiles) ----
    v16h bk[2][2];
    #pragma unroll
    for (int t = 0; t < 2; ++t) {
      const _Float16* krow = Kb + (size_t)(kb + 16 * t + l16) * DHEAD;
      bk[t][0] = *(const v16h*)(krow + 0  + 16 * half);
      bk[t][1] = *(const v16h*)(krow + 32 + 16 * half);
    }

    // ---- per query tile: scores + online softmax + P -> LDS ----
    #pragma unroll
    for (int qq = 0; qq < 2; ++qq) {
      v8f sc[2];
      #pragma unroll
      for (int t = 0; t < 2; ++t) {
        v8f z = {};
        z = wmma16(aq[qq][0], bk[t][0], z);
        z = wmma16(aq[qq][1], bk[t][1], z);
        sc[t] = z;
      }

      float mloc[8];
      #pragma unroll
      for (int r = 0; r < 8; ++r) {
        const int qrow = qbase + 16 * qq + r + 8 * half;
        float s0 = sc[0][r] * scale;
        float s1 = sc[1][r] * scale;
        if (kb + l16      > qrow) s0 = -1.0e30f;
        if (kb + 16 + l16 > qrow) s1 = -1.0e30f;
        sc[0][r] = s0; sc[1][r] = s1;
        mloc[r] = fmaxf(s0, s1);
      }
      // reduce across the 16 lanes sharing each row (xor<16 stays in half)
      #pragma unroll
      for (int mk = 1; mk < 16; mk <<= 1)
        #pragma unroll
        for (int r = 0; r < 8; ++r)
          mloc[r] = fmaxf(mloc[r], __shfl_xor(mloc[r], mk, 32));

      float corr[8];
      #pragma unroll
      for (int r = 0; r < 8; ++r) {
        const float mn = fmaxf(rmax[qq][r], mloc[r]);
        corr[r] = __expf(rmax[qq][r] - mn);
        rmax[qq][r] = mn;
        sc[0][r] = __expf(sc[0][r] - mn);
        sc[1][r] = __expf(sc[1][r] - mn);
      }
      float rs[8];
      #pragma unroll
      for (int r = 0; r < 8; ++r) rs[r] = sc[0][r] + sc[1][r];
      #pragma unroll
      for (int mk = 1; mk < 16; mk <<= 1)
        #pragma unroll
        for (int r = 0; r < 8; ++r)
          rs[r] += __shfl_xor(rs[r], mk, 32);
      #pragma unroll
      for (int r = 0; r < 8; ++r)
        rsum[qq][r] = rsum[qq][r] * corr[r] + rs[r];
      #pragma unroll
      for (int t = 0; t < 4; ++t)
        #pragma unroll
        for (int r = 0; r < 8; ++r) ctx[qq][t][r] *= corr[r];

      // P: C-layout -> LDS (f16)
      #pragma unroll
      for (int r = 0; r < 8; ++r) {
        const int row = r + 8 * half;
        Plds[w][qq][row][l16]      = (_Float16)sc[0][r];
        Plds[w][qq][row][l16 + 16] = (_Float16)sc[1][r];
      }
    }

    asm volatile("s_wait_dscnt 0" ::: "memory");  // same-wave LDS RAW

    // P A-fragments (one per query tile)
    v16h ap[2];
    #pragma unroll
    for (int qq = 0; qq < 2; ++qq) {
      ap[qq] = *(const v16h*)&Plds[w][qq][l16][16 * half];
    }

    // ---- ctx += P x V : V B-frags shared by both query tiles ----
    // Vt layout [dh][s]: b[i] = V[kb+16h+i][dh] = Vt[dh][kb+16h+i] contiguous
    #pragma unroll
    for (int t = 0; t < 4; ++t) {
      const _Float16* vrow = Vb + (size_t)(16 * t + l16) * S_LEN + kb + 16 * half;
      v16h bv = *(const v16h*)vrow;
      ctx[0][t] = wmma16(ap[0], bv, ctx[0][t]);
      ctx[1][t] = wmma16(ap[1], bv, ctx[1][t]);
    }
  }

  // ---- normalize, store ctx as f16 [B,S,D] ----
  const int head = bh & (NHEAD - 1);
  const int bidx = bh >> 4;
  #pragma unroll
  for (int qq = 0; qq < 2; ++qq)
    #pragma unroll
    for (int t = 0; t < 4; ++t)
      #pragma unroll
      for (int r = 0; r < 8; ++r) {
        const int srow = qbase + 16 * qq + r + 8 * half;
        const int col  = head * DHEAD + 16 * t + l16;
        const float vv = ctx[qq][t][r] / rsum[qq][r];
        ctxOut[((size_t)bidx * S_LEN + srow) * DMODEL + col] = (_Float16)vv;
      }
}

// ---------------------------------------------------------------------------
// Output projection: out[m,n] = sum_k ctx[m,k]*Wp[n,k] + bp[n], fp32 out.
// ---------------------------------------------------------------------------
__global__ __launch_bounds__(256) void outproj_kernel(
    const _Float16* __restrict__ Xh, const float* __restrict__ W,
    const float* __restrict__ bias, float* __restrict__ out)
{
  const int lane = threadIdx.x & 31;
  const int w    = threadIdx.x >> 5;
  const int wm   = w & 1, wn = w >> 1;
  const int half = lane >> 4, l16 = lane & 15;
  const int mbase = blockIdx.x * 128 + wm * 64;
  const int nbase = blockIdx.y * 128 + wn * 32;

  v8f acc[4][2];
  #pragma unroll
  for (int nt = 0; nt < 2; ++nt) {
    float bv = bias[nbase + nt * 16 + l16];
    #pragma unroll
    for (int mt = 0; mt < 4; ++mt)
      #pragma unroll
      for (int r = 0; r < 8; ++r)
        acc[mt][nt][r] = bv;
  }

  for (int kb = 0; kb < DMODEL; kb += 32) {
    v16h a[4];
    #pragma unroll
    for (int mt = 0; mt < 4; ++mt) {
      const _Float16* xp = Xh + (size_t)(mbase + mt * 16 + l16) * DMODEL + kb;
      a[mt] = *(const v16h*)(xp + 16 * half);
    }
    v16h b[2];
    #pragma unroll
    for (int nt = 0; nt < 2; ++nt) {
      const float* wp = W + (size_t)(nbase + nt * 16 + l16) * DMODEL + kb + 16 * half;
      b[nt] = load16f_cvt(wp);
    }
    #pragma unroll
    for (int mt = 0; mt < 4; ++mt)
      #pragma unroll
      for (int nt = 0; nt < 2; ++nt)
        acc[mt][nt] = wmma16(a[mt], b[nt], acc[mt][nt]);
  }

  #pragma unroll
  for (int mt = 0; mt < 4; ++mt)
    #pragma unroll
    for (int nt = 0; nt < 2; ++nt) {
      const int n = nbase + nt * 16 + l16;
      #pragma unroll
      for (int r = 0; r < 8; ++r) {
        const int m = mbase + mt * 16 + r + 8 * half;
        out[(size_t)m * DMODEL + n] = acc[mt][nt][r];
      }
    }
}

// ---------------------------------------------------------------------------
extern "C" void kernel_launch(void* const* d_in, const int* in_sizes, int n_in,
                              void* d_out, int out_size, void* d_ws, size_t ws_size,
                              hipStream_t stream) {
  const float* q  = (const float*)d_in[0];
  const float* k  = (const float*)d_in[1];
  const float* v  = (const float*)d_in[2];
  // d_in[3] = causal tril mask -> applied analytically in-kernel
  const float* Wq = (const float*)d_in[4];
  const float* bq = (const float*)d_in[5];
  const float* Wk = (const float*)d_in[6];
  const float* bk = (const float*)d_in[7];
  const float* Wv = (const float*)d_in[8];
  const float* bv = (const float*)d_in[9];
  const float* Wp = (const float*)d_in[10];
  const float* bp = (const float*)d_in[11];
  float* out = (float*)d_out;

  char* ws = (char*)d_ws;
  const size_t MB = 1u << 20;
  _Float16* Qh = (_Float16*)(ws + 0 * MB);    // [B,H,S,DH]   8 MB
  _Float16* Kh = (_Float16*)(ws + 8 * MB);    // [B,H,S,DH]   8 MB
  _Float16* Vt = (_Float16*)(ws + 16 * MB);   // [B,H,DH,S]   8 MB
  _Float16* Ch = (_Float16*)(ws + 24 * MB);   // [B,S,D]      8 MB

  dim3 gProj(32, 8), bProj(256);
  hipLaunchKernelGGL(proj_f16_kernel, gProj, bProj, 0, stream, q, Wq, bq, Qh, 0);
  hipLaunchKernelGGL(proj_f16_kernel, gProj, bProj, 0, stream, k, Wk, bk, Kh, 0);
  hipLaunchKernelGGL(proj_f16_kernel, gProj, bProj, 0, stream, v, Wv, bv, Vt, 1);
  hipLaunchKernelGGL(attn_flash_kernel, dim3(512), dim3(128), 0, stream,
                     Qh, Kh, Vt, Ch);
  hipLaunchKernelGGL(outproj_kernel, gProj, bProj, 0, stream, Ch, Wp, bp, out);
}
